// ScaledDotProductAttention_73770358276772
// MI455X (gfx1250) — matrix operand order repaired and checked by reading, not compiled, not run on hardware
//
#include <hip/hip_runtime.h>

#define B_  2
#define H_  16
#define S_  2048
#define DH_ 64

typedef __attribute__((ext_vector_type(16))) __bf16 v16bf;
typedef __attribute__((ext_vector_type(8)))  float  v8f;
typedef __attribute__((ext_vector_type(4)))  int    v4i;

union BF16x16 { v16bf v; __bf16 b[16]; };

#if __has_builtin(__builtin_amdgcn_global_load_async_to_lds_b128)
#define HAS_ASYNC_LDS 1
typedef __attribute__((address_space(1))) v4i g_v4i;   // global-AS int4
typedef __attribute__((address_space(3))) v4i l_v4i;   // LDS-AS int4
#else
#define HAS_ASYNC_LDS 0
#endif

__device__ __forceinline__ void wait_async_zero() {
#if HAS_ASYNC_LDS
#if __has_builtin(__builtin_amdgcn_s_wait_asynccnt)
    __builtin_amdgcn_s_wait_asynccnt(0);
#else
    asm volatile("s_wait_asynccnt 0" ::: "memory");
#endif
#endif
}

// ---------------------------------------------------------------------------
// Kernel 1: raw scores = (Q K^T) * 1/sqrt(Dh) + bias, masked keys -> -1e30.
// One block per (q-tile of 16 rows, b*H+h); 4 waves, each wave owns every
// 4th key tile. QK^T via v_wmma_f32_16x16x32_bf16 (f32 accumulate).
// ---------------------------------------------------------------------------
__global__ void __launch_bounds__(128)
attn_scores_kernel(const float* __restrict__ q, const float* __restrict__ kmat,
                   const float* __restrict__ bias, const unsigned char* __restrict__ kpm,
                   float* __restrict__ attn) {
    const int qt   = blockIdx.x;       // 16-row q tile
    const int bh   = blockIdx.y;       // b*H + h
    const int b    = bh / H_;
    const int lane = threadIdx.x & 31;
    const int wave = threadIdx.x >> 5;
    const int m    = lane & 15;
    const int half = lane >> 4;
    const float scale = 0.125f;        // 1/sqrt(64)

    // A-matrix (Q tile, 16x32 bf16) fragments for kb = 0,32 -- loop invariant.
    // Layout: lanes 0-15 hold K=e (e<8) / K=e+8 (e>=8); lanes 16-31 add +8.
    const float* qrow = q + ((size_t)bh * S_ + (size_t)qt * 16 + m) * DH_;
    BF16x16 aq[2];
#pragma unroll
    for (int kb = 0; kb < 2; ++kb) {
#pragma unroll
        for (int e = 0; e < 16; ++e) {
            const int kk = (e < 8 ? e : e + 8) + 8 * half + kb * 32;
            aq[kb].b[e] = (__bf16)qrow[kk];      // native v_cvt bf16 path
        }
    }

    for (int kt = wave; kt < S_ / 16; kt += 4) {
        // B-matrix (K^T tile, 32x16 bf16): lane's column n = key row kt*16+n.
        // Lanes 0-15 hold K=e (e<8) / K=e+8 (e>=8); lanes 16-31 add +8
        // (same order as the A fragment), from key row at kb*32.
        const int krow = kt * 16 + m;
        v8f acc = {};
#pragma unroll
        for (int kb = 0; kb < 2; ++kb) {
            const float* kptr = kmat + ((size_t)bh * S_ + krow) * DH_ + kb * 32;
            BF16x16 bk;
#pragma unroll
            for (int e = 0; e < 16; ++e) bk.b[e] = (__bf16)kptr[(e < 8 ? e : e + 8) + 8 * half];
            acc = __builtin_amdgcn_wmma_f32_16x16x32_bf16(
                false, aq[kb].v, false, bk.v, (short)0, acc, false, false);
        }

        // Epilogue: scale + bias + key-padding mask, write raw scores.
        const int  ncol   = kt * 16 + m;                 // this lane's column
        const bool masked = kpm[b * S_ + ncol] != 0;
#pragma unroll
        for (int r = 0; r < 8; ++r) {
            const int qrow_i = qt * 16 + r + 8 * half;   // C/D layout row
            float s = acc[r] * scale + bias[(size_t)qrow_i * S_ + ncol];
            s = masked ? -1.0e30f : s;
            attn[((size_t)bh * S_ + qrow_i) * S_ + ncol] = s;
        }
    }
}

// ---------------------------------------------------------------------------
// Kernel 2: in-place row softmax over the attn region. One block per row of
// 2048; 256 threads x 8 contiguous elements; LDS tree reductions.
// ---------------------------------------------------------------------------
__global__ void __launch_bounds__(256)
softmax_rows_kernel(float* __restrict__ attn) {
    float* p = attn + (size_t)blockIdx.x * S_;
    const int t = threadIdx.x;

    float vals[8];
    float m = -3.0e38f;
#pragma unroll
    for (int i = 0; i < 8; ++i) { vals[i] = p[t * 8 + i]; m = fmaxf(m, vals[i]); }

    __shared__ float red[256];
    red[t] = m;
    __syncthreads();
    for (int off = 128; off > 0; off >>= 1) {
        if (t < off) red[t] = fmaxf(red[t], red[t + off]);
        __syncthreads();
    }
    m = red[0];
    __syncthreads();

    float sum = 0.0f;
#pragma unroll
    for (int i = 0; i < 8; ++i) { vals[i] = __expf(vals[i] - m); sum += vals[i]; }
    red[t] = sum;
    __syncthreads();
    for (int off = 128; off > 0; off >>= 1) {
        if (t < off) red[t] += red[t + off];
        __syncthreads();
    }
    const float inv = 1.0f / red[0];
#pragma unroll
    for (int i = 0; i < 8; ++i) p[t * 8 + i] = vals[i] * inv;
}

// ---------------------------------------------------------------------------
// Kernel 3: out = attn @ V. One block per (q-tile16, b*H+h); 4 waves, wave w
// owns output columns [16w, 16w+16). V tile (32x64) staged in LDS via the
// gfx1250 async memory->LDS path (GLOBAL_LOAD_ASYNC_TO_LDS_B128, ASYNCcnt),
// bf16 WMMA with f32 accumulate over K = 2048.
// ---------------------------------------------------------------------------
__global__ void __launch_bounds__(128)
attn_v_kernel(const float* __restrict__ attn, const float* __restrict__ vmat,
              float* __restrict__ out) {
    const int qt   = blockIdx.x;
    const int bh   = blockIdx.y;
    const int lane = threadIdx.x & 31;
    const int wave = threadIdx.x >> 5;
    const int m    = lane & 15;
    const int half = lane >> 4;
    const int d0   = wave * 16;

    __shared__ float vtile[32 * DH_];    // 8 KB

    const float* arow = attn + ((size_t)bh * S_ + (size_t)qt * 16 + m) * S_;
    v8f acc = {};

    for (int kt = 0; kt < S_ / 32; ++kt) {
        __syncthreads();                 // previous iteration done reading LDS
        // Stage V tile: 32 rows x 64 cols = 512 float4, 4 per thread.
        {
            const float4* s4 = (const float4*)(vmat + ((size_t)bh * S_ + (size_t)kt * 32) * DH_);
            float4* d4 = (float4*)vtile;
#pragma unroll
            for (int i = 0; i < 4; ++i) {
                const int idx = threadIdx.x + i * 128;
#if HAS_ASYNC_LDS
                __builtin_amdgcn_global_load_async_to_lds_b128(
                    (g_v4i*)(s4 + idx), (l_v4i*)(d4 + idx), 0, 0);
#else
                d4[idx] = s4[idx];
#endif
            }
            wait_async_zero();           // ASYNCcnt -> 0 before the barrier
        }
        if (kt + 1 < S_ / 32)            // hint next attn strip into cache
            __builtin_prefetch(arow + (size_t)(kt + 1) * 32, 0, 0);
        __syncthreads();

        // A fragment: attn row m, K slice [kt*32, kt*32+32), f32 -> bf16.
        BF16x16 ap;
#pragma unroll
        for (int e = 0; e < 16; ++e) {
            const int kk = (e < 8 ? e : e + 8) + 8 * half;
            ap.b[e] = (__bf16)arow[(size_t)kt * 32 + kk];
        }
        // B fragment: V tile column d0+m, K rows (e<8 ? e : e+8) + 8*half (from LDS).
        BF16x16 bv;
#pragma unroll
        for (int e = 0; e < 16; ++e)
            bv.b[e] = (__bf16)vtile[((e < 8 ? e : e + 8) + 8 * half) * DH_ + d0 + m];

        acc = __builtin_amdgcn_wmma_f32_16x16x32_bf16(
            false, ap.v, false, bv.v, (short)0, acc, false, false);
    }

#pragma unroll
    for (int r = 0; r < 8; ++r) {
        const int qrow_i = qt * 16 + r + 8 * half;
        out[((size_t)bh * S_ + qrow_i) * DH_ + d0 + m] = acc[r];
    }
}

// ---------------------------------------------------------------------------
extern "C" void kernel_launch(void* const* d_in, const int* in_sizes, int n_in,
                              void* d_out, int out_size, void* d_ws, size_t ws_size,
                              hipStream_t stream) {
    (void)in_sizes; (void)n_in; (void)out_size; (void)d_ws; (void)ws_size;

    const float*         q    = (const float*)d_in[0];
    const float*         k    = (const float*)d_in[1];
    const float*         v    = (const float*)d_in[2];
    const float*         bias = (const float*)d_in[3];
    const unsigned char* kpm  = (const unsigned char*)d_in[4];  // numpy bool_ = 1 byte

    float* out  = (float*)d_out;
    float* attn = out + (size_t)B_ * H_ * S_ * DH_;   // tuple: (out, attn) flat

    const dim3 blk(128);
    const dim3 grd(S_ / 16, B_ * H_);

    attn_scores_kernel<<<grd, blk, 0, stream>>>(q, k, bias, kpm, attn);
    softmax_rows_kernel<<<dim3(B_ * H_ * S_), dim3(256), 0, stream>>>(attn);
    attn_v_kernel<<<grd, blk, 0, stream>>>(attn, v, out);
}
